// Attention_24867860644314
// MI455X (gfx1250) — matrix operand order repaired and checked by reading, not compiled, not run on hardware
//
#include <hip/hip_runtime.h>

// ---------------------------------------------------------------------------
// Local-window attention block (timm-style local_attention), MI455X / gfx1250.
//   B=4, N=197, DIM=2048, H=32, hd=64, WINDOW=8.
// Pipeline:
//   1) gemm_bf16<0>: qkv = x @ qkv_w^T   (WMMA bf16, f32 accum) -> q(*0.125),k,v
//   2) local_attn : windowed masked softmax-attention (wave32 VALU + shuffles)
//   3) gemm_bf16<1>: out = attn_out @ proj_w^T + proj_b (WMMA bf16)
// GEMM: 128x128 workgroup tile, 8 waves each owning 32x64 (8 WMMA / K-step),
// double-buffered LDS, register-staged global->bf16 pipeline, 1 barrier/step.
// ---------------------------------------------------------------------------

#define BSZ     4
#define N_TOK   197
#define DIM_C   2048
#define H_HEADS 32
#define HD      64
#define M_ROWS  (BSZ * N_TOK)          // 788

typedef __attribute__((ext_vector_type(16))) __bf16 bf16x16;
typedef __attribute__((ext_vector_type(8)))  __bf16 bf16x8;
typedef __attribute__((ext_vector_type(8)))  float  f32x8;

#define BM 128
#define BN 128
#define BK 32
#define LDT 40   // LDS row stride (elems): 80 B keeps 16 B alignment, spreads banks

static __device__ __forceinline__ bf16x8 cvt8(const float4 f0, const float4 f1) {
  bf16x8 v;
  v[0]=(__bf16)f0.x; v[1]=(__bf16)f0.y; v[2]=(__bf16)f0.z; v[3]=(__bf16)f0.w;
  v[4]=(__bf16)f1.x; v[5]=(__bf16)f1.y; v[6]=(__bf16)f1.z; v[7]=(__bf16)f1.w;
  return v;
}

// C = A(MxK,f32) @ Bw(NxK,f32)^T ; f32->bf16 conversion fused into staging.
// MODE 0: scatter columns into q/k/v (B,H,N,hd), scale q by 0.125.
// MODE 1: add bias, store plain MxN f32.
template <int MODE>
__global__ __launch_bounds__(256) void gemm_bf16(
    const float* __restrict__ A,
    const float* __restrict__ Bw,
    const float* __restrict__ bias,
    float* __restrict__ out0,   // MODE0: q   MODE1: out
    float* __restrict__ out1,   // MODE0: k
    float* __restrict__ out2,   // MODE0: v
    int M, int N, int K)
{
  __shared__ __align__(16) __bf16 As[2][BM * LDT];
  __shared__ __align__(16) __bf16 Bs[2][BN * LDT];

  const int tid  = threadIdx.x;
  const int lane = tid & 31;
  const int wave = tid >> 5;
  const int wm   = wave >> 1;          // 0..3  (32-row strip)
  const int wn   = wave & 1;           // 0..1  (64-col half)
  const int l16  = lane & 15;
  const int half = lane >> 4;

  const int m0 = blockIdx.x * BM;
  const int n0 = blockIdx.y * BN;

  f32x8 acc[2][4] = {};

  // staging: each thread owns 16 floats of A and 16 of B per K-step
  const int sr = tid >> 1;             // 0..127  tile row
  const int sc = (tid & 1) * 16;       // 0,16    tile col
  const int agrow   = m0 + sr;
  const bool arow_ok = (agrow < M);
  const float* aptr = A  + (size_t)(arow_ok ? agrow : 0) * K + sc;
  const float* bptr = Bw + (size_t)(n0 + sr) * K + sc;

  bf16x8 ra0, ra1, rb0, rb1;

  auto gload = [&](int k0) {
    bf16x8 z = {};
    ra0 = z; ra1 = z;
    if (arow_ok) {
      const float4* ap = (const float4*)(aptr + k0);
      ra0 = cvt8(ap[0], ap[1]);
      ra1 = cvt8(ap[2], ap[3]);
    }
    const float4* bp = (const float4*)(bptr + k0);
    rb0 = cvt8(bp[0], bp[1]);
    rb1 = cvt8(bp[2], bp[3]);
    if (k0 + BK < K) {
      __builtin_prefetch(aptr + k0 + BK, 0, 1);
      __builtin_prefetch(bptr + k0 + BK, 0, 1);
    }
  };

  gload(0);
  const int nk = K / BK;

  for (int kt = 0; kt < nk; ++kt) {
    const int cur = kt & 1;
    // commit staged registers to current LDS buffer
    *(bf16x8*)(&As[cur][sr * LDT + sc])     = ra0;
    *(bf16x8*)(&As[cur][sr * LDT + sc + 8]) = ra1;
    *(bf16x8*)(&Bs[cur][sr * LDT + sc])     = rb0;
    *(bf16x8*)(&Bs[cur][sr * LDT + sc + 8]) = rb1;
    __syncthreads();

    // kick off next tile's global loads to overlap with WMMA below
    if (kt + 1 < nk) gload((kt + 1) * BK);

    // ---- load ALL fragments first (distinct regs -> single dscnt wait),
    //      then issue the 8-WMMA chain back-to-back.
    // A frag (16x32): lane l16 -> row M=l16; lanes 0-15 K 0..7 & 16..23,
    // lanes 16-31 K 8..15 & 24..31.
    bf16x16 af[2];
#pragma unroll
    for (int mi = 0; mi < 2; ++mi) {
      const int arow = wm * 32 + mi * 16 + l16;
      const bf16x8 a0 = *(const bf16x8*)(&As[cur][arow * LDT + half * 8]);
      const bf16x8 a1 = *(const bf16x8*)(&As[cur][arow * LDT + half * 8 + 16]);
      af[mi] = __builtin_shufflevector(a0, a1,
               0,1,2,3,4,5,6,7,8,9,10,11,12,13,14,15);
    }
    // B frag (32x16): lane -> column N=l16; lanes 0-15 K 0..7 & 16..23,
    // lanes 16-31 K 8..15 & 24..31 (matches A frag).
    bf16x16 bfr[4];
#pragma unroll
    for (int j = 0; j < 4; ++j) {
      const int brow = wn * 64 + j * 16 + l16;
      const bf16x8 b0 = *(const bf16x8*)(&Bs[cur][brow * LDT + half * 8]);
      const bf16x8 b1 = *(const bf16x8*)(&Bs[cur][brow * LDT + half * 8 + 16]);
      bfr[j] = __builtin_shufflevector(b0, b1,
               0,1,2,3,4,5,6,7,8,9,10,11,12,13,14,15);
    }
#pragma unroll
    for (int mi = 0; mi < 2; ++mi)
#pragma unroll
      for (int j = 0; j < 4; ++j)
        acc[mi][j] = __builtin_amdgcn_wmma_f32_16x16x32_bf16(
            false, af[mi], false, bfr[j], (short)0, acc[mi][j], false, false);
    // next iteration writes the other buffer; the dscnt-draining sync above
    // (next loop head) protects the ping-pong.
  }

  // ---- epilogue: C layout VGPR r -> (M = r + half*8, N = l16) ----
#pragma unroll
  for (int mi = 0; mi < 2; ++mi) {
#pragma unroll
    for (int j = 0; j < 4; ++j) {
      const int gcol = n0 + wn * 64 + j * 16 + l16;
      if (MODE == 0) {
        const int qi  = gcol >> 11;        // 0=q 1=k 2=v
        const int rem = gcol & 2047;
        const int h   = rem >> 6;
        const int d   = rem & 63;
        float* dst = (qi == 0) ? out0 : (qi == 1) ? out1 : out2;
        const float sc2 = (qi == 0) ? 0.125f : 1.0f;   // fold hd^-0.5 into q
#pragma unroll
        for (int r = 0; r < 8; ++r) {
          const int grow = m0 + wm * 32 + mi * 16 + half * 8 + r;
          if (grow < M) {
            const int bb = grow / N_TOK;
            const int n  = grow - bb * N_TOK;
            dst[(((size_t)bb * H_HEADS + h) * N_TOK + n) * HD + d] =
                acc[mi][j][r] * sc2;
          }
        }
      } else {
        const float bv = bias[gcol];
#pragma unroll
        for (int r = 0; r < 8; ++r) {
          const int grow = m0 + wm * 32 + mi * 16 + half * 8 + r;
          if (grow < M)
            out0[(size_t)grow * N + gcol] = acc[mi][j][r] + bv;
        }
      }
    }
  }
}

// One wave32 per (b,h,i) row. Window j in [i-8, i+8), mask weight
// w = (8 - |i-j|/2)/8; softmax over the <=16 in-window scores; out = p @ v.
__global__ __launch_bounds__(256) void local_attn(
    const float* __restrict__ qb, const float* __restrict__ kb,
    const float* __restrict__ vb, float* __restrict__ ao)
{
  const int lane = threadIdx.x & 31;
  const int rid  = blockIdx.x * 8 + (threadIdx.x >> 5);
  if (rid >= BSZ * H_HEADS * N_TOK) return;

  const int i  = rid % N_TOK;
  const int bh = rid / N_TOK;
  const int h  = bh % H_HEADS;
  const int b  = bh / H_HEADS;

  const float* qrow = qb + (size_t)rid * HD;   // q pre-scaled by 0.125

  const int  jj    = lane & 15;                // window position 0..15
  const int  j     = i - 8 + jj;               // j in [i-8, i+7]
  const bool valid = (j >= 0) && (j < N_TOK);

  // scores: pair (lane, lane^16) splits the 64-d dot product
  float part = 0.0f;
  if (valid) {
    const float* krow = kb + (size_t)(rid - i + j) * HD;
    const int d0 = (lane >> 4) * 32;
#pragma unroll
    for (int d = 0; d < 32; ++d) part += qrow[d0 + d] * krow[d0 + d];
  }
  part += __shfl_xor(part, 16, 32);            // full dot, replicated in halves

  const int   dist = (jj >= 8) ? (jj - 8) : (8 - jj);
  const float w    = 1.0f - 0.0625f * (float)dist;   // (8 - dist/2)/8
  const float s    = valid ? part * w : -3.0e38f;

  float mval = s;
#pragma unroll
  for (int off = 1; off <= 8; off <<= 1) mval = fmaxf(mval, __shfl_xor(mval, off, 32));
  float p = valid ? __expf(s - mval) : 0.0f;
  float sum = p;
#pragma unroll
  for (int off = 1; off <= 8; off <<= 1) sum += __shfl_xor(sum, off, 32);
  const float rinv = 1.0f / sum;

  // out[d] = sum_j p_j * v[j][d]; lane owns d = lane and lane+32
  float acc0 = 0.0f, acc1 = 0.0f;
#pragma unroll
  for (int t = 0; t < 16; ++t) {
    const int jt = i - 8 + t;
    if (jt >= 0 && jt < N_TOK) {
      const float pt = __shfl(p, t, 32);
      const float* vrow = vb + (size_t)(rid - i + jt) * HD;
      acc0 += pt * vrow[lane];
      acc1 += pt * vrow[lane + 32];
    }
  }
  float* orow = ao + ((size_t)(b * N_TOK + i)) * DIM_C + h * HD;
  orow[lane]      = acc0 * rinv;
  orow[lane + 32] = acc1 * rinv;
}

extern "C" void kernel_launch(void* const* d_in, const int* in_sizes, int n_in,
                              void* d_out, int out_size, void* d_ws, size_t ws_size,
                              hipStream_t stream) {
  const float* x      = (const float*)d_in[0];   // (4,197,2048)
  const float* qkv_w  = (const float*)d_in[1];   // (6144,2048)
  const float* proj_w = (const float*)d_in[2];   // (2048,2048)
  const float* proj_b = (const float*)d_in[3];   // (2048,)
  float* out = (float*)d_out;                    // (4,197,2048)

  const size_t seg = (size_t)BSZ * H_HEADS * N_TOK * HD;  // 1,613,824 floats
  float* ws = (float*)d_ws;
  float* qb = ws;
  float* kb = ws + seg;
  float* vb = ws + 2 * seg;
  float* ao = ws + 3 * seg;                      // (788, 2048) attn output

  const int mtiles = (M_ROWS + BM - 1) / BM;     // 7
  dim3 blk(256);

  // 1) QKV projection + head-split scatter (q scaled)
  gemm_bf16<0><<<dim3(mtiles, (3 * DIM_C) / BN), blk, 0, stream>>>(
      x, qkv_w, nullptr, qb, kb, vb, M_ROWS, 3 * DIM_C, DIM_C);

  // 2) local windowed attention
  local_attn<<<(BSZ * H_HEADS * N_TOK) / 8, blk, 0, stream>>>(qb, kb, vb, ao);

  // 3) output projection + bias
  gemm_bf16<1><<<dim3(mtiles, DIM_C / BN), blk, 0, stream>>>(
      ao, proj_w, proj_b, out, nullptr, nullptr, M_ROWS, DIM_C, DIM_C);
}
